// mofnet_14465449853405
// MI455X (gfx1250) — matrix operand order repaired and checked by reading, not compiled, not run on hardware
//
#include <hip/hip_runtime.h>
#include <hip/hip_bf16.h>

typedef __attribute__((ext_vector_type(16))) __bf16 v16bf;
typedef __attribute__((ext_vector_type(8)))  __bf16 v8bf;
typedef __attribute__((ext_vector_type(8)))  float  v8f;
typedef __attribute__((ext_vector_type(4)))  unsigned int v4u;
typedef __attribute__((ext_vector_type(8)))  int v8i;
typedef __attribute__((ext_vector_type(4)))  int v4i;

#define TE     128
#define COLS   49152          // 3*128*128 == 768*64
#define NBLK   384            // COLS / 128
#define WELEMS (TE * COLS)    // 6291456 elements per weight matrix
#define TOPO_N 3000
#define EPS    1e-5f

// ------------------------------------------------------------------
// Native CDNA5 tanh (V_TANH_F32 is a 1-cycle-issue TRANS op that
// co-executes with the WMMA pipe).  Branch-free fallback if the
// builtin name differs on this toolchain.
// ------------------------------------------------------------------
__device__ __forceinline__ float fast_tanh(float x) {
#if __has_builtin(__builtin_amdgcn_tanhf)
    return __builtin_amdgcn_tanhf(x);
#elif __has_builtin(__builtin_amdgcn_tanh_f32)
    return __builtin_amdgcn_tanh_f32(x);
#else
    // tanh(x) = 1 - 2/(exp2(2*log2(e)*x) + 1), monotone, saturates correctly
    float e = __builtin_amdgcn_exp2f(x * 2.885390082f);
    return 1.f - 2.f * __builtin_amdgcn_rcpf(e + 1.f);
#endif
}

// ------------------------------------------------------------------
// TDM: async-load one contiguous 8KB slice (4096 bf16) global -> LDS.
// D# built per CDNA5 ISA (group0 128b, group1 256b).
// This toolchain exposes the 6-arg builtin (extra int32x8 group).
// ------------------------------------------------------------------
__device__ __forceinline__ void tdm_slice_load(const __bf16* gsrc, unsigned lds_byte) {
    unsigned long long ga = (unsigned long long)gsrc;
    v4u g0;
    g0[0] = 1u;                                              // count=1, user mode
    g0[1] = lds_byte;                                        // lds_addr
    g0[2] = (unsigned)(ga & 0xffffffffu);                    // global_addr[31:0]
    g0[3] = (unsigned)((ga >> 32) & 0x1ffffffu) | (2u << 30);// addr[56:32] | type=2
    v8i g1;
    g1[0] = (int)(1u << 16);                 // workgroup_mask=0, data_size=1 (2B)
    g1[1] = (int)((4096u & 0xffffu) << 16);  // tensor_dim0 lo16 @ bits[63:48]
    g1[2] = (int)((4096u >> 16) | (1u << 16)); // tensor_dim0 hi16, tensor_dim1=1 lo16
    g1[3] = (int)(4096u << 16);              // tensor_dim1 hi16=0, tile_dim0=4096
    g1[4] = (int)(1u);                       // tile_dim1=1, tile_dim2=0
    g1[5] = (int)(4096u);                    // tensor_dim0_stride lo32
    g1[6] = 0;                               // stride hi, dim1_stride lo
    g1[7] = 0;
    v4i gz4 = {0, 0, 0, 0};
    v8i gz8 = {0, 0, 0, 0, 0, 0, 0, 0};
    __builtin_amdgcn_tensor_load_to_lds(g0, g1, gz4, gz4, gz8, 0);
}

// ------------------------------------------------------------------
// Prep: f32 -> bf16 conversion; weights transposed to [col][TE] so a
// 32-column slice of a 128-col block is one contiguous 8KB TDM tile.
// grid (1536, 4) x 256 threads.  y==3 converts topo_table (no transpose).
// ------------------------------------------------------------------
__global__ void __launch_bounds__(256) prep_kernel(
    const float* __restrict__ nW, const float* __restrict__ eW,
    const float* __restrict__ iW, const float* __restrict__ topo,
    __bf16* __restrict__ nWT, __bf16* __restrict__ eWT,
    __bf16* __restrict__ iWT, __bf16* __restrict__ topoB) {
    if (blockIdx.y == 3) {
        int stride = gridDim.x * 256;
        for (int i = blockIdx.x * 256 + threadIdx.x; i < TOPO_N * TE; i += stride)
            topoB[i] = (__bf16)topo[i];
        return;
    }
    const float* src = (blockIdx.y == 0) ? nW : (blockIdx.y == 1 ? eW : iW);
    __bf16*      dst = (blockIdx.y == 0) ? nWT : (blockIdx.y == 1 ? eWT : iWT);
    __shared__ float sh[32 * 130];     // padded to dodge bank conflicts
    const int col0 = blockIdx.x * 32;
    const int tid  = threadIdx.x;
    for (int idx = tid; idx < 4096; idx += 256) {        // coalesced read
        int r = idx >> 5, c = idx & 31;
        sh[c * 130 + r] = src[(size_t)r * COLS + col0 + c];
    }
    __syncthreads();
    for (int idx = tid; idx < 4096; idx += 256) {        // coalesced write
        int c = idx >> 7, r = idx & 127;
        dst[(size_t)(col0 + c) * TE + r] = (__bf16)sh[c * 130 + r];
    }
}

// ------------------------------------------------------------------
// Fused hypernet kernel: 1 WG = 16 samples, 4 waves, wave w owns
// columns [32w,32w+32) of each 128-col block via private double-buffered
// TDM staging -> zero barriers in the steady-state GEMM loops.
// ------------------------------------------------------------------
__global__ void __launch_bounds__(128) mofnet_kernel(
    const int*   __restrict__ xg,
    const float* __restrict__ node_table, const float* __restrict__ edge_table,
    const float* __restrict__ nbias, const float* __restrict__ ebias, const float* __restrict__ ibias,
    const float* __restrict__ n_g, const float* __restrict__ n_b, const float* __restrict__ n_m, const float* __restrict__ n_v,
    const float* __restrict__ e_g, const float* __restrict__ e_b, const float* __restrict__ e_m, const float* __restrict__ e_v,
    const float* __restrict__ i_g, const float* __restrict__ i_b, const float* __restrict__ i_m, const float* __restrict__ i_v,
    const float* __restrict__ pW,
    const float* __restrict__ p_g, const float* __restrict__ p_b, const float* __restrict__ p_m, const float* __restrict__ p_v,
    const float* __restrict__ oW, const float* __restrict__ ob,
    const __bf16* __restrict__ nWT, const __bf16* __restrict__ eWT, const __bf16* __restrict__ iWT,
    const __bf16* __restrict__ topoB,
    float* __restrict__ dout) {

    __shared__ __align__(128) __bf16 Bstage[4][2][4096]; // 64KB: per-wave double buffers
    __shared__ float tbl_lds[16 * 384];                  // 24KB: ne (phase0) / ee (phase1)
    __shared__ float cat_lds[16 * 768];                  // 48KB: raw h|g, then BN'd cat
    __shared__ float m_lds[16 * 64];                     // 4KB
    __shared__ float p_lds[16 * 32];                     // 2KB

    const int tid   = threadIdx.x;
    const int lane  = tid & 31;
    const int w     = tid >> 5;       // wave id 0..3 (wave32)
    const int cil   = lane & 15;      // column-in-tile / row id
    const int khalf = lane >> 4;      // K-half selector per WMMA layout
    const int base  = blockIdx.x * 16;

    // ---- zero accumulators ----
    for (int i = tid; i < 16 * 768; i += 128) cat_lds[i] = 0.f;
    for (int i = tid; i < 16 * 64;  i += 128) m_lds[i]  = 0.f;

    // ---- A fragments: topo rows in wave32 WMMA 16x32 bf16 layout, resident all kernel ----
    const int topoIdx = xg[(base + cil) * 7];
    const __bf16* rowp = topoB + (size_t)topoIdx * TE;
    v16bf afrag[4];
#pragma unroll
    for (int kb = 0; kb < 4; ++kb) {
        v8bf lo = *(const v8bf*)(rowp + kb * 32 + khalf * 16);       // K {0-7 | 16-23}
        v8bf hi = *(const v8bf*)(rowp + kb * 32 + khalf * 16 + 8);   // K {8-15 | 24-31}
#pragma unroll
        for (int i = 0; i < 8; ++i) { afrag[kb][i] = lo[i]; afrag[kb][i + 8] = hi[i]; }
    }

    const unsigned ldsBase = (unsigned)(unsigned long long)(void*)&Bstage[w][0][0];

    // ================= phases 0 (node) and 1 (edge) =================
    for (int p = 0; p < 2; ++p) {
        const __bf16* WT   = p ? eWT : nWT;
        const float*  bias = p ? ebias : nbias;
        const float*  tabl = p ? edge_table : node_table;

        __syncthreads();                       // tbl_lds free for re-gather
        for (int rw = tid; rw < 48; rw += 128) {
            int s = rw / 3, jx = rw % 3;
            int iv = xg[(base + s) * 7 + 1 + p * 3 + jx];
            if (iv < 0) iv = 0;
            const float4* src = (const float4*)(tabl + (size_t)iv * 128);
            float4*       dst = (float4*)(tbl_lds + s * 384 + jx * 128);
#pragma unroll 4
            for (int q = 0; q < 32; ++q) dst[q] = src[q];
        }
        __syncthreads();

        // TDM pipeline prologue: blocks 0,1 into the two private buffers
        tdm_slice_load(WT + (size_t)w * 4096,          ldsBase);
        tdm_slice_load(WT + (size_t)w * 4096 + 16384,  ldsBase + 8192);

        for (int b = 0; b < NBLK; ++b) {
            // bias for both N-tiles: issue early so latency hides under WMMA
            const float bv0 = bias[b * 128 + w * 32 + cil];
            const float bv1 = bias[b * 128 + w * 32 + 16 + cil];
            if (b + 1 < NBLK) __builtin_amdgcn_s_wait_tensorcnt(1);  // oldest retired
            else              __builtin_amdgcn_s_wait_tensorcnt(0);
            const __bf16* Bl = &Bstage[w][b & 1][0];
            const int j = b >> 7, kcol = b & 127;

            float part[8];
#pragma unroll
            for (int r = 0; r < 8; ++r) part[r] = 0.f;

#pragma unroll
            for (int t = 0; t < 2; ++t) {
                // load all K fragments up front, then chain the WMMAs
                v16bf bf[4];
#pragma unroll
                for (int kb = 0; kb < 4; ++kb)
                    bf[kb] = *(const v16bf*)(Bl + (t * 16 + cil) * 128 + kb * 32 + khalf * 16);
                v8f acc = {};
#pragma unroll
                for (int kb = 0; kb < 4; ++kb)
                    acc = __builtin_amdgcn_wmma_f32_16x16x32_bf16(
                        false, afrag[kb], false, bf[kb], (short)0, acc, false, false);

                const int   ncol = w * 32 + t * 16 + cil;    // l index within block
                const float bv   = t ? bv1 : bv0;
#pragma unroll
                for (int r = 0; r < 8; ++r) {
                    float val = fast_tanh(acc[r] + bv);      // fused hyper-weight elem
                    part[r] += val * tbl_lds[(r + 8 * khalf) * 384 + j * 128 + ncol];
                }
            }
            // reduce 16 lanes -> per-sample partial, accumulate across 4 waves
#pragma unroll
            for (int r = 0; r < 8; ++r) {
                float v = part[r];
                v += __shfl_xor(v, 1); v += __shfl_xor(v, 2);
                v += __shfl_xor(v, 4); v += __shfl_xor(v, 8);
                if (cil == 0)
                    atomicAdd(&cat_lds[(r + 8 * khalf) * 768 + p * 384 + j * 128 + kcol], v);
            }
            if (b + 2 < NBLK)
                tdm_slice_load(WT + (size_t)w * 4096 + (size_t)(b + 2) * 16384,
                               ldsBase + (unsigned)(b & 1) * 8192);
        }
    }

    // ---- BN + ReLU + validity mask -> cat ----
    __syncthreads();
    for (int i = tid; i < 16 * 768; i += 128) {
        int s = i / 768, c = i % 768;
        float vv = cat_lds[i], gg, bb, mm, vr; int iv;
        if (c < 384) {
            gg = n_g[c]; bb = n_b[c]; mm = n_m[c]; vr = n_v[c];
            iv = xg[(base + s) * 7 + 1 + c / 128];
        } else {
            int cc = c - 384;
            gg = e_g[cc]; bb = e_b[cc]; mm = e_m[cc]; vr = e_v[cc];
            iv = xg[(base + s) * 7 + 4 + cc / 128];
        }
        float r = (vv - mm) * rsqrtf(vr + EPS) * gg + bb;
        r = fmaxf(r, 0.f);
        cat_lds[i] = (iv < 0) ? 0.f : r;
    }
    __syncthreads();

    // ================= phase 2 (inter): m[s][k] accumulated in registers =================
    {
        const int jpar = w >> 1;                  // waves 0,1 -> even j, waves 2,3 -> odd j
        float macc[2][8];
#pragma unroll
        for (int t = 0; t < 2; ++t)
#pragma unroll
            for (int r = 0; r < 8; ++r) macc[t][r] = 0.f;

        tdm_slice_load(iWT + (size_t)w * 4096,         ldsBase);
        tdm_slice_load(iWT + (size_t)w * 4096 + 16384, ldsBase + 8192);

        for (int b = 0; b < NBLK; ++b) {
            const float bv0 = ibias[b * 128 + w * 32 + cil];
            const float bv1 = ibias[b * 128 + w * 32 + 16 + cil];
            if (b + 1 < NBLK) __builtin_amdgcn_s_wait_tensorcnt(1);
            else              __builtin_amdgcn_s_wait_tensorcnt(0);
            const __bf16* Bl = &Bstage[w][b & 1][0];
            const int jj = 2 * b + jpar;
            float catv[8];
#pragma unroll
            for (int r = 0; r < 8; ++r) catv[r] = cat_lds[(r + 8 * khalf) * 768 + jj];

#pragma unroll
            for (int t = 0; t < 2; ++t) {
                v16bf bf[4];
#pragma unroll
                for (int kb = 0; kb < 4; ++kb)
                    bf[kb] = *(const v16bf*)(Bl + (t * 16 + cil) * 128 + kb * 32 + khalf * 16);
                v8f acc = {};
#pragma unroll
                for (int kb = 0; kb < 4; ++kb)
                    acc = __builtin_amdgcn_wmma_f32_16x16x32_bf16(
                        false, afrag[kb], false, bf[kb], (short)0, acc, false, false);

                const float bv = t ? bv1 : bv0;
#pragma unroll
                for (int r = 0; r < 8; ++r)
                    macc[t][r] += fast_tanh(acc[r] + bv) * catv[r];
            }
            if (b + 2 < NBLK)
                tdm_slice_load(iWT + (size_t)w * 4096 + (size_t)(b + 2) * 16384,
                               ldsBase + (unsigned)(b & 1) * 8192);
        }
#pragma unroll
        for (int t = 0; t < 2; ++t)
#pragma unroll
            for (int r = 0; r < 8; ++r)
                atomicAdd(&m_lds[(r + 8 * khalf) * 64 + (w & 1) * 32 + t * 16 + cil],
                          macc[t][r]);
    }

    // ================= head: BN(m) -> relu -> @pW -> BN -> relu -> @oW + ob =================
    __syncthreads();
    for (int i = tid; i < 16 * 64; i += 128) {
        int ch = i & 63;
        float r = (m_lds[i] - i_m[ch]) * rsqrtf(i_v[ch] + EPS) * i_g[ch] + i_b[ch];
        m_lds[i] = fmaxf(r, 0.f);
    }
    __syncthreads();
    for (int o = tid; o < 16 * 32; o += 128) {
        int s = o >> 5, po = o & 31;
        float a = 0.f;
#pragma unroll 8
        for (int kk = 0; kk < 64; ++kk) a += m_lds[s * 64 + kk] * pW[kk * 32 + po];
        a = (a - p_m[po]) * rsqrtf(p_v[po] + EPS) * p_g[po] + p_b[po];
        p_lds[o] = fmaxf(a, 0.f);
    }
    __syncthreads();
    if (tid < 16) {
        float a = ob[0];
#pragma unroll
        for (int po = 0; po < 32; ++po) a += p_lds[tid * 32 + po] * oW[po];
        dout[base + tid] = a;
    }
}

// ------------------------------------------------------------------
extern "C" void kernel_launch(void* const* d_in, const int* in_sizes, int n_in,
                              void* d_out, int out_size, void* d_ws, size_t ws_size,
                              hipStream_t stream) {
    (void)in_sizes; (void)n_in; (void)out_size; (void)ws_size;
    const int*   xg         = (const int*)d_in[0];
    const float* topo_table = (const float*)d_in[1];
    const float* node_table = (const float*)d_in[2];
    const float* edge_table = (const float*)d_in[3];
    const float* nW         = (const float*)d_in[4];
    const float* nbias      = (const float*)d_in[5];
    const float* eW         = (const float*)d_in[6];
    const float* ebias      = (const float*)d_in[7];
    const float* iW         = (const float*)d_in[8];
    const float* ibias      = (const float*)d_in[9];
    const float* n_g = (const float*)d_in[10], *n_b = (const float*)d_in[11];
    const float* n_m = (const float*)d_in[12], *n_v = (const float*)d_in[13];
    const float* e_g = (const float*)d_in[14], *e_b = (const float*)d_in[15];
    const float* e_m = (const float*)d_in[16], *e_v = (const float*)d_in[17];
    const float* i_g = (const float*)d_in[18], *i_b = (const float*)d_in[19];
    const float* i_m = (const float*)d_in[20], *i_v = (const float*)d_in[21];
    const float* pW  = (const float*)d_in[22];
    const float* p_g = (const float*)d_in[23], *p_b = (const float*)d_in[24];
    const float* p_m = (const float*)d_in[25], *p_v = (const float*)d_in[26];
    const float* oW  = (const float*)d_in[27];
    const float* ob  = (const float*)d_in[28];

    char* ws = (char*)d_ws;
    __bf16* nWT   = (__bf16*)(ws);
    __bf16* eWT   = (__bf16*)(ws + (size_t)WELEMS * 2);
    __bf16* iWT   = (__bf16*)(ws + (size_t)WELEMS * 4);
    __bf16* topoB = (__bf16*)(ws + (size_t)WELEMS * 6);

    prep_kernel<<<dim3(1536, 4), dim3(256), 0, stream>>>(
        nW, eW, iW, topo_table, nWT, eWT, iWT, topoB);

    mofnet_kernel<<<dim3(128), dim3(128), 0, stream>>>(
        xg, node_table, edge_table, nbias, ebias, ibias,
        n_g, n_b, n_m, n_v, e_g, e_b, e_m, e_v, i_g, i_b, i_m, i_v,
        pW, p_g, p_b, p_m, p_v, oW, ob,
        nWT, eWT, iWT, topoB, (float*)d_out);
}
